// ExpertsModel_17523466568300
// MI455X (gfx1250) — matrix operand order repaired and checked by reading, not compiled, not run on hardware
//
#include <hip/hip_runtime.h>
#include <hip/hip_bf16.h>
#include <stddef.h>

// Problem constants (from the reference).
#define EE 8
#define DD 512
#define FF 2048
#define NN 4096
#define NPAD (NN + EE * 32)      // 4352: every expert segment padded to 32 rows
#define ROWTILES (NPAD / 32)     // 136 macro row-tiles (32 rows each)

typedef __attribute__((ext_vector_type(16))) __bf16 v16bf;
typedef __attribute__((ext_vector_type(8)))  __bf16 v8bf;
typedef __attribute__((ext_vector_type(8)))  float  v8f;

// ---------------------------------------------------------------------------
// Fragment loaders (layouts per CDNA5 ISA 7.12.2, wave32), bf16 sources.
// A 16x32 (MxK), row-major, leading dim `ld` (elements):
//   lane<16:  M=lane,    K = {k0+0..7,  k0+16..23}
//   lane>=16: M=lane-16, K = {k0+8..15, k0+24..31}
// Both 8-half groups are 16B-aligned contiguous -> two b128 loads.
__device__ __forceinline__ v16bf load_a_frag(const __bf16* base, int ld,
                                             int row0, int k0) {
    int lane = threadIdx.x & 31;
    int m    = lane & 15;
    int kb   = (lane < 16) ? 0 : 8;
    const __bf16* p = base + (size_t)(row0 + m) * ld + (k0 + kb);
    v8bf lo = *(const v8bf*)(p);
    v8bf hi = *(const v8bf*)(p + 16);
    return __builtin_shufflevector(lo, hi, 0, 1, 2, 3, 4, 5, 6, 7,
                                           8, 9, 10, 11, 12, 13, 14, 15);
}

// B 32x16 (KxN) from an N-major (transposed) bf16 slab: Wt[n][k], ld = K.
//   lane<16:  N=lane,    K = {k0+0..7,  k0+16..23}
//   lane>=16: N=lane-16, K = {k0+8..15, k0+24..31}
// Two 8-half groups per lane -> two b128 loads.
__device__ __forceinline__ v16bf load_b_frag(const __bf16* baseT, int ldk,
                                             int k0, int col0) {
    int lane = threadIdx.x & 31;
    int n    = lane & 15;
    int kb   = (lane < 16) ? 0 : 8;
    const __bf16* p = baseT + (size_t)(col0 + n) * ldk + (k0 + kb);
    v8bf lo = *(const v8bf*)(p);
    v8bf hi = *(const v8bf*)(p + 16);
    return __builtin_shufflevector(lo, hi, 0, 1, 2, 3, 4, 5, 6, 7,
                                           8, 9, 10, 11, 12, 13, 14, 15);
}

// ---------------------------------------------------------------------------
// 0) Re-init workspace control state (deterministic across graph replays).
__global__ void moe_init_kernel(int* __restrict__ perm,
                                int* __restrict__ counts,
                                int* __restrict__ cursor) {
    int i = blockIdx.x * blockDim.x + threadIdx.x;
    if (i < NPAD) perm[i] = -1;
    if (i < EE) { counts[i] = 0; cursor[i] = 0; }
}

// 1) Gate: one wave per token; lane-split dot products, shfl reduction, argmax.
__global__ void moe_gate_kernel(const float* __restrict__ x,
                                const float* __restrict__ Wg,
                                const float* __restrict__ bg,
                                int* __restrict__ eid,
                                int* __restrict__ counts) {
    int wave = threadIdx.x >> 5;
    int lane = threadIdx.x & 31;
    int t    = blockIdx.x * (blockDim.x >> 5) + wave;
    if (t >= NN) return;

    float acc[EE];
#pragma unroll
    for (int e = 0; e < EE; ++e) acc[e] = 0.f;

    for (int d = lane; d < DD; d += 32) {
        float xv = x[(size_t)t * DD + d];
#pragma unroll
        for (int e = 0; e < EE; ++e) acc[e] += xv * Wg[(size_t)d * EE + e];
    }
#pragma unroll
    for (int e = 0; e < EE; ++e) {
        float v = acc[e];
        for (int off = 16; off > 0; off >>= 1) v += __shfl_xor(v, off, 32);
        acc[e] = v;
    }
    if (lane == 0) {
        int   best = 0;
        float bv   = acc[0] + bg[0];
#pragma unroll
        for (int e = 1; e < EE; ++e) {
            float v = acc[e] + bg[e];
            if (v > bv) { bv = v; best = e; }   // strict > : first-max, like argmax
        }
        eid[t] = best;
        atomicAdd(&counts[best], 1);
    }
}

// 2) Tiny exclusive scan; each expert segment padded to a multiple of 32 rows.
__global__ void moe_scan_kernel(const int* __restrict__ counts,
                                int* __restrict__ poff) {
    int acc = 0;
#pragma unroll
    for (int e = 0; e < EE; ++e) {
        poff[e] = acc;
        acc += ((counts[e] + 31) >> 5) << 5;
    }
    poff[EE] = acc;
}

// 3) Scatter tokens into their expert's padded segment.
__global__ void moe_scatter_kernel(const int* __restrict__ eid,
                                   const int* __restrict__ poff,
                                   int* __restrict__ cursor,
                                   int* __restrict__ perm) {
    int t = blockIdx.x * blockDim.x + threadIdx.x;
    if (t >= NN) return;
    int e   = eid[t];
    int pos = poff[e] + atomicAdd(&cursor[e], 1);
    perm[pos] = t;
}

// 4) Gather x rows (fp32) into compacted bf16 activation buffer; pad rows = 0.
__global__ void moe_gather_kernel(const float* __restrict__ x,
                                  const int* __restrict__ perm,
                                  __bf16* __restrict__ xg) {
    int r = blockIdx.x;
    int t = perm[r];
    for (int c = threadIdx.x; c < DD; c += blockDim.x)
        xg[(size_t)r * DD + c] = (t >= 0) ? (__bf16)x[(size_t)t * DD + c]
                                          : (__bf16)0.f;
}

// 5) Per-expert weight slab transpose + fp32->bf16 convert.
//    src: [z][R][C] fp32  ->  dst: [z][C][R] bf16.  R, C multiples of 32.
__global__ void moe_transpose_kernel(const float* __restrict__ src,
                                     __bf16* __restrict__ dst,
                                     int R, int C) {
    __shared__ float tile[32][33];              // +1 pad: no bank conflicts
    size_t slab = (size_t)R * C;
    const float* s = src + (size_t)blockIdx.z * slab;
    __bf16*      d = dst + (size_t)blockIdx.z * slab;
    int c0 = blockIdx.x * 32, r0 = blockIdx.y * 32;
    int tx = threadIdx.x & 31, ty = threadIdx.x >> 5;   // 8 row-phases
#pragma unroll
    for (int j = 0; j < 4; ++j)
        tile[ty + 8 * j][tx] = s[(size_t)(r0 + ty + 8 * j) * C + c0 + tx];
    __syncthreads();
#pragma unroll
    for (int j = 0; j < 4; ++j)
        d[(size_t)(c0 + ty + 8 * j) * R + r0 + tx] = (__bf16)tile[tx][ty + 8 * j];
}

// ---------------------------------------------------------------------------
// 6) FFN layer 1: hg = relu(xg @ W1[e] + b1[e]).
//    One 32x32 output macro-tile per wave (2x2 register blocking).
__global__ void moe_ffn1_kernel(const __bf16* __restrict__ xg,
                                const __bf16* __restrict__ w1t,   // [E][F][D]
                                const float* __restrict__ b1,
                                const int* __restrict__ poff,
                                __bf16* __restrict__ hg) {
    int wave = threadIdx.x >> 5;
    int lane = threadIdx.x & 31;
    int row0 = blockIdx.x * 32;
    int col0 = (blockIdx.y * 8 + wave) * 32;

    int e = 0;
#pragma unroll
    for (int i = 0; i < EE; ++i)
        if (row0 >= poff[i + 1]) e = i + 1;
    if (e >= EE) e = EE - 1;   // all-pad tile beyond last segment: junk, never used

    const __bf16* Wt = w1t + (size_t)e * FF * DD;
    v8f acc00 = {0.f,0.f,0.f,0.f,0.f,0.f,0.f,0.f};
    v8f acc01 = acc00, acc10 = acc00, acc11 = acc00;
    for (int k0 = 0; k0 < DD; k0 += 32) {
        v16bf a0 = load_a_frag(xg, DD, row0,      k0);
        v16bf a1 = load_a_frag(xg, DD, row0 + 16, k0);
        v16bf b0 = load_b_frag(Wt, DD, k0, col0);
        v16bf b1f = load_b_frag(Wt, DD, k0, col0 + 16);
        acc00 = __builtin_amdgcn_wmma_f32_16x16x32_bf16(false, a0, false, b0,  (short)0, acc00, false, false);
        acc01 = __builtin_amdgcn_wmma_f32_16x16x32_bf16(false, a0, false, b1f, (short)0, acc01, false, false);
        acc10 = __builtin_amdgcn_wmma_f32_16x16x32_bf16(false, a1, false, b0,  (short)0, acc10, false, false);
        acc11 = __builtin_amdgcn_wmma_f32_16x16x32_bf16(false, a1, false, b1f, (short)0, acc11, false, false);
    }
    int n     = lane & 15;
    int mbase = (lane < 16) ? 0 : 8;
    v8f accs[2][2] = {{acc00, acc01}, {acc10, acc11}};
#pragma unroll
    for (int mt = 0; mt < 2; ++mt) {
#pragma unroll
        for (int nt = 0; nt < 2; ++nt) {
            float bias = b1[(size_t)e * FF + col0 + nt * 16 + n];
#pragma unroll
            for (int r = 0; r < 8; ++r) {
                float v = accs[mt][nt][r] + bias;
                v = (v > 0.f) ? v : 0.f;
                hg[(size_t)(row0 + mt * 16 + mbase + r) * FF + col0 + nt * 16 + n]
                    = (__bf16)v;
            }
        }
    }
}

// 7) FFN layer 2: out[token] = hg @ W2[e] + b2[e], scattered via perm.
__global__ void moe_ffn2_kernel(const __bf16* __restrict__ hg,
                                const __bf16* __restrict__ w2t,   // [E][D][F]
                                const float* __restrict__ b2,
                                const int* __restrict__ poff,
                                const int* __restrict__ perm,
                                float* __restrict__ out) {
    int wave = threadIdx.x >> 5;
    int lane = threadIdx.x & 31;
    int row0 = blockIdx.x * 32;
    int col0 = (blockIdx.y * 8 + wave) * 32;

    int e = 0;
#pragma unroll
    for (int i = 0; i < EE; ++i)
        if (row0 >= poff[i + 1]) e = i + 1;
    if (e >= EE) e = EE - 1;

    const __bf16* Wt = w2t + (size_t)e * DD * FF;
    v8f acc00 = {0.f,0.f,0.f,0.f,0.f,0.f,0.f,0.f};
    v8f acc01 = acc00, acc10 = acc00, acc11 = acc00;
    for (int k0 = 0; k0 < FF; k0 += 32) {
        v16bf a0 = load_a_frag(hg, FF, row0,      k0);
        v16bf a1 = load_a_frag(hg, FF, row0 + 16, k0);
        v16bf b0 = load_b_frag(Wt, FF, k0, col0);
        v16bf b1f = load_b_frag(Wt, FF, k0, col0 + 16);
        acc00 = __builtin_amdgcn_wmma_f32_16x16x32_bf16(false, a0, false, b0,  (short)0, acc00, false, false);
        acc01 = __builtin_amdgcn_wmma_f32_16x16x32_bf16(false, a0, false, b1f, (short)0, acc01, false, false);
        acc10 = __builtin_amdgcn_wmma_f32_16x16x32_bf16(false, a1, false, b0,  (short)0, acc10, false, false);
        acc11 = __builtin_amdgcn_wmma_f32_16x16x32_bf16(false, a1, false, b1f, (short)0, acc11, false, false);
    }
    int n     = lane & 15;
    int mbase = (lane < 16) ? 0 : 8;
    v8f accs[2][2] = {{acc00, acc01}, {acc10, acc11}};
#pragma unroll
    for (int mt = 0; mt < 2; ++mt) {
#pragma unroll
        for (int nt = 0; nt < 2; ++nt) {
            float bias = b2[(size_t)e * DD + col0 + nt * 16 + n];
#pragma unroll
            for (int r = 0; r < 8; ++r) {
                int tok = perm[row0 + mt * 16 + mbase + r];
                if (tok >= 0)
                    out[(size_t)tok * DD + col0 + nt * 16 + n]
                        = accs[mt][nt][r] + bias;
            }
        }
    }
}

// ---------------------------------------------------------------------------
extern "C" void kernel_launch(void* const* d_in, const int* in_sizes, int n_in,
                              void* d_out, int out_size, void* d_ws, size_t ws_size,
                              hipStream_t stream) {
    const float* x  = (const float*)d_in[0];   // [N, D]
    const float* Wg = (const float*)d_in[1];   // [D, E]
    const float* bg = (const float*)d_in[2];   // [E]
    const float* W1 = (const float*)d_in[3];   // [E, D, F]
    const float* b1 = (const float*)d_in[4];   // [E, F]
    const float* W2 = (const float*)d_in[5];   // [E, F, D]
    const float* b2 = (const float*)d_in[6];   // [E, D]
    float* out = (float*)d_out;                // [N, D]

    // Workspace layout (~56 MB total), 512B-aligned regions.
    char* ws = (char*)d_ws;
    size_t off = 0;
    auto take = [&](size_t bytes) { void* p = ws + off;
                                    off = (off + bytes + 511) & ~(size_t)511;
                                    return p; };
    int*    eid    = (int*)take(NN * sizeof(int));
    int*    counts = (int*)take(EE * sizeof(int));
    int*    cursor = (int*)take(EE * sizeof(int));
    int*    poff   = (int*)take((EE + 1) * sizeof(int));
    int*    perm   = (int*)take(NPAD * sizeof(int));
    __bf16* xg     = (__bf16*)take((size_t)NPAD * DD * sizeof(__bf16));
    __bf16* hg     = (__bf16*)take((size_t)NPAD * FF * sizeof(__bf16));
    __bf16* w1t    = (__bf16*)take((size_t)EE * DD * FF * sizeof(__bf16));
    __bf16* w2t    = (__bf16*)take((size_t)EE * FF * DD * sizeof(__bf16));
    (void)ws_size; (void)in_sizes; (void)n_in; (void)out_size;

    moe_init_kernel<<<(NPAD + 255) / 256, 256, 0, stream>>>(perm, counts, cursor);
    moe_gate_kernel<<<NN / 8, 256, 0, stream>>>(x, Wg, bg, eid, counts);
    moe_scan_kernel<<<1, 1, 0, stream>>>(counts, poff);
    moe_scatter_kernel<<<NN / 256, 256, 0, stream>>>(eid, poff, cursor, perm);
    moe_gather_kernel<<<NPAD, 256, 0, stream>>>(x, perm, xg);
    // W1: [E][D][F] -> w1t [E][F][D];  W2: [E][F][D] -> w2t [E][D][F]
    moe_transpose_kernel<<<dim3(FF / 32, DD / 32, EE), 256, 0, stream>>>(W1, w1t, DD, FF);
    moe_transpose_kernel<<<dim3(DD / 32, FF / 32, EE), 256, 0, stream>>>(W2, w2t, FF, DD);
    moe_ffn1_kernel<<<dim3(ROWTILES, FF / 256), 256, 0, stream>>>(xg, w1t, b1, poff, hg);
    moe_ffn2_kernel<<<dim3(ROWTILES, DD / 256), 256, 0, stream>>>(hg, w2t, b2, poff, perm, out);
}
